// MultiModalEmbedding_87832081204089
// MI455X (gfx1250) — matrix operand order repaired and checked by reading, not compiled, not run on hardware
//
#include <hip/hip_runtime.h>
#include <hip/hip_bf16.h>

typedef __attribute__((ext_vector_type(16))) _Float16 v16h;
typedef __attribute__((ext_vector_type(8)))  float    v8f;
typedef __attribute__((ext_vector_type(4)))  int      v4i_t;

#define DMODEL 1024

#if defined(__has_builtin)
#if __has_builtin(__builtin_amdgcn_global_load_async_to_lds_b128)
#define ASYNC_LDS 1
#endif
#endif
#ifndef ASYNC_LDS
#define ASYNC_LDS 0
#endif

typedef __attribute__((address_space(1))) v4i_t* as1_v4i;
typedef __attribute__((address_space(3))) v4i_t* as3_v4i;

union Frag16 { v16h h; unsigned int u[8]; };

__device__ __forceinline__ void async_wait0() {
#if ASYNC_LDS
#if defined(__has_builtin) && __has_builtin(__builtin_amdgcn_s_wait_asynccnt)
    __builtin_amdgcn_s_wait_asynccnt(0);
#else
    asm volatile("s_wait_asynccnt 0x0" ::: "memory");
#endif
#endif
}

// ---------------- prep kernels ----------------

__global__ void k_f32_to_f16(const float* __restrict__ in, _Float16* __restrict__ out, int n) {
    int i = blockIdx.x * 256 + threadIdx.x;
    if (i < n) out[i] = (_Float16)in[i];
}

// out (Rout x W) f16 from in (Rin x W) f32; rows >= Rin are zero (K-pad weights)
__global__ void k_pad_rows(const float* __restrict__ in, _Float16* __restrict__ out,
                           int Rin, int Rout, int W) {
    int i = blockIdx.x * 256 + threadIdx.x;
    if (i >= Rout * W) return;
    int r = i / W;
    int c = i - r * W;
    out[i] = (r < Rin) ? (_Float16)in[r * W + c] : (_Float16)0.0f;
}

// out (R x Kout) f16 from in (R x Kin) f32; cols >= Kin are zero (K-pad activations)
__global__ void k_pad_cols(const float* __restrict__ in, _Float16* __restrict__ out,
                           int R, int Kin, int Kout) {
    int i = blockIdx.x * 256 + threadIdx.x;
    if (i >= R * Kout) return;
    int r = i / Kout;
    int k = i - r * Kout;
    out[i] = (k < Kin) ? (_Float16)in[r * Kin + k] : (_Float16)0.0f;
}

// maps (B,64,21,21) f32 -> patches (B*49, 576) f16 with feature order (s1, s2, c)
__global__ void k_patches(const float* __restrict__ maps, _Float16* __restrict__ A, int total) {
    int i = blockIdx.x * 256 + threadIdx.x;
    if (i >= total) return;
    int f = i % 576;
    int r = i / 576;
    int b = r / 49, p = r - b * 49;
    int h = p / 7,  w = p - h * 7;
    int c   = f & 63;
    int s12 = f >> 6;
    int s1  = s12 / 3, s2 = s12 - 3 * s1;
    A[i] = (_Float16)maps[((size_t)(b * 64 + c) * 21 + 3 * h + s1) * 21 + (3 * w + s2)];
}

// 2D sinusoidal PE table: pe (T x 1024). xy==nullptr -> patch-center mode.
__global__ void k_pe(const float* __restrict__ xy, int stride, float* __restrict__ pe, int T) {
    int i = blockIdx.x * 256 + threadIdx.x;
    if (i >= T * DMODEL) return;
    int t = i >> 10;
    int d = i & 1023;
    float x, y;
    if (xy) { x = xy[t * stride]; y = xy[t * stride + 1]; }
    else    { int h = t / 7, w = t - 7 * h; x = (float)(3 * h + 1); y = (float)(3 * w + 1); }
    float v  = (d < 512) ? x : y;
    int   dd = d & 511;
    int   fi = dd >> 1;
    float f = __expf(-9.210340371976184f * ((float)fi * (1.0f / 256.0f)));   // 1e-4^(fi/256)
    float a = v * f;
    pe[i] = (dd & 1) ? __cosf(a) : __sinf(a);
}

// ---------------- WMMA GEMM ----------------
// C(M x 1024) = A(M x K, lda) @ W(K x 1024), f32 accum via v_wmma_f32_16x16x32_f16.
// Double-buffered LDS; A tile via GLOBAL_LOAD_ASYNC_TO_LDS_B128 (ASYNCcnt), B tile
// software-pipelined: global b128 issued before compute, pack+ds_store committed after.
// mode 0: outH[m*1024+n] = f16(relu(acc + bias[n]))
// mode 1: outF[(b*82 + rowBase + t)*1024 + n] = acc + bias[n] + pe[t*1024+n]
__global__ __launch_bounds__(256) void k_gemm(
    const _Float16* __restrict__ A, int K, int lda,
    const _Float16* __restrict__ W,
    const float* __restrict__ bias,
    _Float16* __restrict__ outH,
    float* __restrict__ outF,
    const float* __restrict__ pe,
    int tokensPerBatch, int rowBase, int mode)
{
    // A: 128 rows x 32 halfs, row stride 20 dwords (16B-aligned rows, conflict-free frag reads)
    // B: 32 K x 128 N, paired-K dword layout: BsDw[(k>>1)*128 + n] = {k even lo16, k odd hi16}
    __shared__ unsigned int AsDw[2][128 * 20];
    __shared__ unsigned int BsDw[2][16 * 128];

    const int tid  = threadIdx.x;
    const int lane = tid & 31;
    const int wid  = tid >> 5;
    const int wm   = wid & 3;      // 4 waves along M (32 rows each)
    const int wn   = wid >> 2;     // 2 waves along N (64 cols each)
    const int lm   = lane & 15;
    const int lh   = lane >> 4;    // half-wave select per ISA fragment layout
    const int m0   = blockIdx.y * 128;
    const int n0   = blockIdx.x * 128;

    v8f acc[2][4] = {};

    // in-flight staging registers
    uint4 rB0, rB1;
#if !ASYNC_LDS
    uint4 rA0, rA1;
#endif
    const int kpB = tid >> 4;                 // B K-pair 0..15
    const int cB  = (tid & 15) * 8;           // B col block

    auto issueA = [&](int kk) {               // A tile: 2 x 16B per thread
        #pragma unroll
        for (int it = 0; it < 2; ++it) {
            int chunk = tid + it * 256;       // 0..511 (8 halfs each)
            int row   = chunk >> 2;
            int part  = chunk & 3;
            const _Float16* gp = A + (size_t)(m0 + row) * lda + kk + part * 8;
#if ASYNC_LDS
            int buf = (kk >> 5) & 1;
            unsigned int* lp = &AsDw[buf][row * 20 + part * 4];
            __builtin_amdgcn_global_load_async_to_lds_b128((as1_v4i)gp, (as3_v4i)lp, 0, 0);
#else
            uint4 v = *reinterpret_cast<const uint4*>(gp);
            if (it == 0) rA0 = v; else rA1 = v;
#endif
        }
    };
    auto issueB = [&](int kk) {
        rB0 = *reinterpret_cast<const uint4*>(W + (size_t)(kk + 2 * kpB)     * DMODEL + n0 + cB);
        rB1 = *reinterpret_cast<const uint4*>(W + (size_t)(kk + 2 * kpB + 1) * DMODEL + n0 + cB);
    };
    auto commit = [&](int buf) {
#if !ASYNC_LDS
        #pragma unroll
        for (int it = 0; it < 2; ++it) {
            int chunk = tid + it * 256;
            int row   = chunk >> 2;
            int part  = chunk & 3;
            *reinterpret_cast<uint4*>(&AsDw[buf][row * 20 + part * 4]) = (it == 0) ? rA0 : rA1;
        }
#endif
        unsigned int ra[4] = {rB0.x, rB0.y, rB0.z, rB0.w};
        unsigned int rb[4] = {rB1.x, rB1.y, rB1.z, rB1.w};
        unsigned int pk[8];
        #pragma unroll
        for (int i = 0; i < 4; ++i) {
            pk[2 * i]     = (ra[i] & 0xffffu) | (rb[i] << 16);
            pk[2 * i + 1] = (ra[i] >> 16)     | (rb[i] & 0xffff0000u);
        }
        unsigned int* lp = &BsDw[buf][kpB * 128 + cB];
        reinterpret_cast<uint4*>(lp)[0] = make_uint4(pk[0], pk[1], pk[2], pk[3]);
        reinterpret_cast<uint4*>(lp)[1] = make_uint4(pk[4], pk[5], pk[6], pk[7]);
    };

    const int nk = K >> 5;
    issueA(0); issueB(0); commit(0);

    for (int ik = 0; ik < nk; ++ik) {
        const int cur  = ik & 1;
        const bool more = (ik + 1 < nk);
        async_wait0();
        __syncthreads();                      // buf[cur] fully populated (ASYNCcnt + DScnt drained)
        if (more) { issueA((ik + 1) << 5); issueB((ik + 1) << 5); }

        // ---- fragments per ISA 7.12.2 layouts ----
        Frag16 a[2], b[4];
        #pragma unroll
        for (int mt = 0; mt < 2; ++mt) {
            const unsigned int* Ar = &AsDw[cur][(wm * 32 + mt * 16 + lm) * 20];
            #pragma unroll
            for (int j = 0; j < 4; ++j) a[mt].u[j]     = Ar[4 * lh + j];       // K 8lh..8lh+7
            #pragma unroll
            for (int j = 0; j < 4; ++j) a[mt].u[4 + j] = Ar[8 + 4 * lh + j];   // K 16+8lh..
        }
        #pragma unroll
        for (int nt = 0; nt < 4; ++nt) {
            int col = wn * 64 + nt * 16 + lm;
            #pragma unroll
            for (int j = 0; j < 8; ++j) b[nt].u[j] = BsDw[cur][(4 * lh + (j & 3) + 8 * (j >> 2)) * 128 + col];
        }
        #pragma unroll
        for (int mt = 0; mt < 2; ++mt)
            #pragma unroll
            for (int nt = 0; nt < 4; ++nt)
                acc[mt][nt] = __builtin_amdgcn_wmma_f32_16x16x32_f16(
                    false, a[mt].h, false, b[nt].h, (short)0, acc[mt][nt], false, false);

        if (more) commit(cur ^ 1);            // pack+store after compute: load latency hidden
    }

    // ---- epilogue ----
    #pragma unroll
    for (int mt = 0; mt < 2; ++mt) {
        #pragma unroll
        for (int nt = 0; nt < 4; ++nt) {
            int n = n0 + wn * 64 + nt * 16 + lm;
            float bn = bias[n];
            #pragma unroll
            for (int r = 0; r < 8; ++r) {
                int m = m0 + wm * 32 + mt * 16 + r + 8 * lh;   // C-matrix lane layout
                float v = acc[mt][nt][r] + bn;
                if (mode == 0) {
                    outH[(size_t)m * DMODEL + n] = (_Float16)fmaxf(v, 0.0f);
                } else {
                    int bI = m / tokensPerBatch;
                    int t  = m - bI * tokensPerBatch;
                    if (pe) v += pe[t * DMODEL + n];
                    outF[((size_t)bI * 82 + rowBase + t) * DMODEL + n] = v;
                }
            }
        }
    }
}

// ---------------- host ----------------

extern "C" void kernel_launch(void* const* d_in, const int* in_sizes, int n_in,
                              void* d_out, int out_size, void* d_ws, size_t ws_size,
                              hipStream_t stream) {
    const float* maps = (const float*)d_in[0];
    const float* ship = (const float*)d_in[1];
    const float* game = (const float*)d_in[2];
    const float* Wp1  = (const float*)d_in[3];  const float* bp1 = (const float*)d_in[4];
    const float* Wp2  = (const float*)d_in[5];  const float* bp2 = (const float*)d_in[6];
    const float* Ws1  = (const float*)d_in[7];  const float* bs1 = (const float*)d_in[8];
    const float* Ws2  = (const float*)d_in[9];  const float* bs2 = (const float*)d_in[10];
    const float* Wg1  = (const float*)d_in[11]; const float* bg1 = (const float*)d_in[12];
    const float* Wg2  = (const float*)d_in[13]; const float* bg2 = (const float*)d_in[14];
    float* out = (float*)d_out;

    const int B = 1024;
    const int Mp = B * 49, Ms = B * 32, Mg = B;

    char* w = (char*)d_ws;
    auto alloc = [&](size_t bytes) { char* p = w; w += (bytes + 255) & ~(size_t)255; return p; };

    _Float16* Wp1h = (_Float16*)alloc((size_t)576 * 1024 * 2);
    _Float16* Wp2h = (_Float16*)alloc((size_t)1024 * 1024 * 2);
    _Float16* Ws1h = (_Float16*)alloc((size_t)32 * 1024 * 2);
    _Float16* Ws2h = (_Float16*)alloc((size_t)1024 * 1024 * 2);
    _Float16* Wg1h = (_Float16*)alloc((size_t)32 * 1024 * 2);
    _Float16* Wg2h = (_Float16*)alloc((size_t)1024 * 1024 * 2);
    _Float16* Ap   = (_Float16*)alloc((size_t)Mp * 576 * 2);
    _Float16* As   = (_Float16*)alloc((size_t)Ms * 32 * 2);
    _Float16* Ag   = (_Float16*)alloc((size_t)Mg * 32 * 2);
    _Float16* Hp   = (_Float16*)alloc((size_t)Mp * 1024 * 2);
    _Float16* Hs   = (_Float16*)alloc((size_t)Ms * 1024 * 2);
    _Float16* Hg   = (_Float16*)alloc((size_t)Mg * 1024 * 2);
    float*    peP  = (float*)alloc((size_t)49 * 1024 * 4);
    float*    peS  = (float*)alloc((size_t)32 * 1024 * 4);

    auto grid1 = [](int n) { return dim3((unsigned)((n + 255) / 256)); };

    // weight converts / pads
    k_f32_to_f16<<<grid1(576 * 1024), 256, 0, stream>>>(Wp1, Wp1h, 576 * 1024);
    k_f32_to_f16<<<grid1(1024 * 1024), 256, 0, stream>>>(Wp2, Wp2h, 1024 * 1024);
    k_pad_rows  <<<grid1(32 * 1024), 256, 0, stream>>>(Ws1, Ws1h, 8, 32, 1024);
    k_f32_to_f16<<<grid1(1024 * 1024), 256, 0, stream>>>(Ws2, Ws2h, 1024 * 1024);
    k_pad_rows  <<<grid1(32 * 1024), 256, 0, stream>>>(Wg1, Wg1h, 16, 32, 1024);
    k_f32_to_f16<<<grid1(1024 * 1024), 256, 0, stream>>>(Wg2, Wg2h, 1024 * 1024);

    // activation A-matrices
    k_patches <<<grid1(Mp * 576), 256, 0, stream>>>(maps, Ap, Mp * 576);
    k_pad_cols<<<grid1(Ms * 32), 256, 0, stream>>>(ship, As, Ms, 8, 32);
    k_pad_cols<<<grid1(Mg * 32), 256, 0, stream>>>(game, Ag, Mg, 16, 32);

    // positional-encoding tables (batch-invariant)
    k_pe<<<grid1(49 * 1024), 256, 0, stream>>>(nullptr, 0, peP, 49);
    k_pe<<<grid1(32 * 1024), 256, 0, stream>>>(ship, 8, peS, 32);   // batch 0 coords

    // layer 1: relu(A @ W1 + b1) -> f16 hidden
    k_gemm<<<dim3(8, Mp / 128), 256, 0, stream>>>(Ap, 576, 576, Wp1h, bp1, Hp, nullptr, nullptr, 1, 0, 0);
    k_gemm<<<dim3(8, Ms / 128), 256, 0, stream>>>(As, 32, 32, Ws1h, bs1, Hs, nullptr, nullptr, 1, 0, 0);
    k_gemm<<<dim3(8, Mg / 128), 256, 0, stream>>>(Ag, 32, 32, Wg1h, bg1, Hg, nullptr, nullptr, 1, 0, 0);

    // layer 2: H @ W2 + b2 (+PE), scattered into concat layout [emb_s | emb_g | emb_p]
    k_gemm<<<dim3(8, Ms / 128), 256, 0, stream>>>(Hs, 1024, 1024, Ws2h, bs2, nullptr, out, peS, 32, 0, 1);
    k_gemm<<<dim3(8, Mg / 128), 256, 0, stream>>>(Hg, 1024, 1024, Wg2h, bg2, nullptr, out, nullptr, 1, 32, 1);
    k_gemm<<<dim3(8, Mp / 128), 256, 0, stream>>>(Hp, 1024, 1024, Wp2h, bp2, nullptr, out, peP, 49, 33, 1);
}
